// CustomTransformerRegressor_78847009620256
// MI455X (gfx1250) — compile-verified, matrix operand order checked by reading, not run on hardware
//
#include <hip/hip_runtime.h>
#include <hip/hip_bf16.h>

typedef __attribute__((ext_vector_type(16))) _Float16 v16h;
typedef __attribute__((ext_vector_type(8)))  float    v8f;

union Frag { v16h h; unsigned int u[8]; };

// ---- CDNA5 async global->LDS copy (guarded; falls back to VGPR copy) ------
#if defined(__has_builtin)
# if __has_builtin(__builtin_amdgcn_global_load_async_to_lds_b32)
#  define USE_ASYNC_COPY 1
# endif
# if __has_builtin(__builtin_amdgcn_s_wait_asynccnt)
#  define HAVE_WAIT_ASYNC_BUILTIN 1
# endif
#endif
#ifndef USE_ASYNC_COPY
# define USE_ASYNC_COPY 0
#endif
#ifndef HAVE_WAIT_ASYNC_BUILTIN
# define HAVE_WAIT_ASYNC_BUILTIN 0
#endif

typedef __attribute__((address_space(1))) int gbl_i32;
typedef __attribute__((address_space(3))) int lds_i32;

__device__ __forceinline__ void copy_b32_to_lds(const _Float16* g, _Float16* l) {
#if USE_ASYNC_COPY
    __builtin_amdgcn_global_load_async_to_lds_b32(
        (gbl_i32*)(void*)g, (lds_i32*)(void*)l, 0, 0);
#else
    *(unsigned int*)l = *(const unsigned int*)g;
#endif
}

__device__ __forceinline__ void wait_async_copies() {
#if USE_ASYNC_COPY
# if HAVE_WAIT_ASYNC_BUILTIN
    __builtin_amdgcn_s_wait_asynccnt(0);
# else
    asm volatile("s_wait_asynccnt 0x0" ::: "memory");
# endif
#endif
}

// ---------------------------------------------------------------------------
// f32 -> f16 elementwise convert (weights)
// ---------------------------------------------------------------------------
__global__ __launch_bounds__(256) void cvt_f32_f16(const float* __restrict__ in,
                                                   _Float16* __restrict__ out, int n) {
    int i = blockIdx.x * 256 + threadIdx.x;
    if (i < n) out[i] = (_Float16)in[i];
}

// ---------------------------------------------------------------------------
// input projection: x = src @ W_in + b_in  (IN=3 -> D=512), writes f32 + f16
// ---------------------------------------------------------------------------
__global__ __launch_bounds__(256) void input_proj(const float* __restrict__ src,
                                                  const float* __restrict__ Win,
                                                  const float* __restrict__ bin,
                                                  float* __restrict__ xf,
                                                  _Float16* __restrict__ xh,
                                                  int M, int D) {
    int idx = blockIdx.x * 256 + threadIdx.x;
    if (idx >= M * D) return;
    int row = idx / D, d = idx - row * D;
    float s = bin[d];
#pragma unroll
    for (int i = 0; i < 3; ++i) s += src[row * 3 + i] * Win[i * D + d];
    xf[idx] = s;
    xh[idx] = (_Float16)s;
}

// ---------------------------------------------------------------------------
// WMMA GEMM: C[M,N] = epi(A[M,K] @ W[K,N] + bias)
// MODE 0: store f16      MODE 1: relu, store f16      MODE 2: +resid, store f32
// block: 256 thr (8 waves), tile BM=256 x BN=64; each wave owns 32x64
// (2 M-frags x 4 N-frags = 8 WMMAs per 32-K step, B-frags reused 2x)
// ---------------------------------------------------------------------------
template <int MODE>
__global__ __launch_bounds__(256) void gemm_wmma(const _Float16* __restrict__ A,
                                                 const _Float16* __restrict__ W,
                                                 const float* __restrict__ bias,
                                                 const float* __restrict__ resid,
                                                 void* __restrict__ outp,
                                                 int M, int N, int K) {
    constexpr int BM = 256, BN = 64, BK = 32;
    constexpr int AS = BK + 2;  // even pad keeps K-pairs 4B aligned
    constexpr int BS = BK + 2;
    __shared__ _Float16 As[BM * AS];   // [row][k]
    __shared__ _Float16 Bs[BN * BS];   // [n][k]  (transposed so K-pairs contiguous)

    const int tid  = threadIdx.x;
    const int wave = tid >> 5;
    const int lane = tid & 31;
    const int lh   = lane >> 4;
    const int ln16 = lane & 15;
    const int m0 = blockIdx.y * BM;
    const int n0 = blockIdx.x * BN;

    v8f acc[2][4] = {};

    // 16-bit A/B fragment K-offsets for this lane (ISA 7.12.2 layout)
    int kofs[8];
#pragma unroll
    for (int i = 0; i < 8; ++i) kofs[i] = ((i < 4) ? 2 * i : 2 * i + 8) + 8 * lh;

    for (int k0 = 0; k0 < K; k0 += BK) {
        __syncthreads();
        // A tile: 256x32 halves = 4096 dwords, 16/thread, straight copy -> async
#pragma unroll
        for (int i = 0; i < 16; ++i) {
            int e  = tid + 256 * i;            // dword index
            int r  = e >> 4;                   // 16 dwords per row
            int kk = (e & 15) * 2;
            copy_b32_to_lds(A + (size_t)(m0 + r) * K + k0 + kk, &As[r * AS + kk]);
        }
        // B tile: 32x64 halves, stored transposed [n][k] (through VGPRs)
#pragma unroll
        for (int i = 0; i < 8; ++i) {
            int e  = tid + 256 * i;            // half index
            int kk = e >> 6;
            int n  = e & 63;
            Bs[n * BS + kk] = W[(size_t)(k0 + kk) * N + n0 + n];
            if (i == 0 && k0 + BK < K)
                __builtin_prefetch(W + (size_t)(k0 + BK + kk) * N + n0 + n, 0, 1);
        }
        wait_async_copies();
        __syncthreads();

        Frag a0, a1;
#pragma unroll
        for (int i = 0; i < 8; ++i) {
            a0.u[i] = *(const unsigned int*)(&As[(wave * 32 + ln16) * AS + kofs[i]]);
            a1.u[i] = *(const unsigned int*)(&As[(wave * 32 + 16 + ln16) * AS + kofs[i]]);
        }
#pragma unroll
        for (int nt = 0; nt < 4; ++nt) {
            Frag b;
#pragma unroll
            for (int i = 0; i < 8; ++i)
                b.u[i] = *(const unsigned int*)(&Bs[(nt * 16 + ln16) * BS + kofs[i]]);
            acc[0][nt] = __builtin_amdgcn_wmma_f32_16x16x32_f16(
                false, a0.h, false, b.h, (short)0, acc[0][nt], false, false);
            acc[1][nt] = __builtin_amdgcn_wmma_f32_16x16x32_f16(
                false, a1.h, false, b.h, (short)0, acc[1][nt], false, false);
        }
    }

    // epilogue: C 16x16 f32 layout -> row = r + 8*lh, col = lane&15
#pragma unroll
    for (int mt = 0; mt < 2; ++mt) {
#pragma unroll
        for (int nt = 0; nt < 4; ++nt) {
#pragma unroll
            for (int r = 0; r < 8; ++r) {
                int row = m0 + wave * 32 + mt * 16 + r + 8 * lh;
                int col = n0 + nt * 16 + ln16;
                float v = acc[mt][nt][r] + bias[col];
                if constexpr (MODE == 1) v = v > 0.f ? v : 0.f;
                if constexpr (MODE == 2) {
                    float* o = (float*)outp;
                    o[(size_t)row * N + col] = v + resid[(size_t)row * N + col];
                } else {
                    _Float16* o = (_Float16*)outp;
                    o[(size_t)row * N + col] = (_Float16)v;
                }
            }
        }
    }
}

// ---------------------------------------------------------------------------
// flash attention: block = (b, h, 64-query tile); 4 waves, 16 q-rows each;
// streams 64-key tiles (16 WMMAs per tile) with online softmax.
// Q/K/V/O are f16 [B,S,D].
// ---------------------------------------------------------------------------
__global__ __launch_bounds__(128) void flash_attn(const _Float16* __restrict__ Q,
                                                  const _Float16* __restrict__ Km,
                                                  const _Float16* __restrict__ V,
                                                  _Float16* __restrict__ O,
                                                  int S, int D) {
    constexpr int DH = 64, KT = 64;
    constexpr int KS = 66;  // even pad, K-pairs stay 4B aligned
    __shared__ _Float16 Klds[KT * KS];      // [key][dim]
    __shared__ _Float16 Vlds[DH * KS];      // [dim][key]
    __shared__ _Float16 Plds[4 * 16 * KS];  // per-wave P rows [row][key]

    const int tid  = threadIdx.x;
    const int wave = tid >> 5;
    const int lane = tid & 31;
    const int lh   = lane >> 4;
    const int ln16 = lane & 15;
    const int b  = blockIdx.z, h = blockIdx.y;
    const int q0 = blockIdx.x * 64;
    const size_t base = ((size_t)b * S) * D + (size_t)h * DH;
    const float scale = 0.125f;  // 1/sqrt(64)

    int kofs[8];
#pragma unroll
    for (int i = 0; i < 8; ++i) kofs[i] = ((i < 4) ? 2 * i : 2 * i + 8) + 8 * lh;

    // Q fragments: 16x64 = two 16x32 A-frags, loaded straight from global
    Frag qf[2];
    {
        const size_t qrow = base + (size_t)(q0 + wave * 16 + ln16) * D;
#pragma unroll
        for (int t = 0; t < 2; ++t)
#pragma unroll
            for (int i = 0; i < 8; ++i)
                qf[t].u[i] = *(const unsigned int*)(Q + qrow + t * 32 + kofs[i]);
    }

    float mi[8], li[8];
    v8f oacc[4] = {};
#pragma unroll
    for (int r = 0; r < 8; ++r) { mi[r] = -3.0e38f; li[r] = 0.f; }

    const int nkt = S / KT;
    for (int j = 0; j < nkt; ++j) {
        __syncthreads();  // previous iteration's LDS reads complete
        // K tile 64x64 halves = 2048 dwords (straight copy -> async path)
#pragma unroll
        for (int i = 0; i < 16; ++i) {
            int e   = tid + 128 * i;        // dword idx [0,2048)
            int key = e >> 5;
            int dd  = (e & 31) * 2;
            copy_b32_to_lds(Km + base + (size_t)(j * KT + key) * D + dd,
                            &Klds[key * KS + dd]);
        }
        // V tile, transposed to [dim][key] (through VGPRs)
#pragma unroll
        for (int i = 0; i < 32; ++i) {
            int e   = tid + 128 * i;        // half idx [0,4096)
            int key = e >> 6;
            int dd  = e & 63;
            Vlds[dd * KS + key] = V[base + (size_t)(j * KT + key) * D + dd];
        }
        wait_async_copies();
        __syncthreads();

        // S = Q @ K^T : 16x64 scores per wave; 4 C-tiles x 2 K-steps = 8 WMMAs
        v8f s[4] = {};
#pragma unroll
        for (int nt = 0; nt < 4; ++nt) {
            Frag b0, b1;
#pragma unroll
            for (int i = 0; i < 8; ++i) {
                b0.u[i] = *(const unsigned int*)(&Klds[(nt * 16 + ln16) * KS + kofs[i]]);
                b1.u[i] = *(const unsigned int*)(&Klds[(nt * 16 + ln16) * KS + 32 + kofs[i]]);
            }
            s[nt] = __builtin_amdgcn_wmma_f32_16x16x32_f16(false, qf[0].h, false, b0.h,
                                                           (short)0, s[nt], false, false);
            s[nt] = __builtin_amdgcn_wmma_f32_16x16x32_f16(false, qf[1].h, false, b1.h,
                                                           (short)0, s[nt], false, false);
        }

        // online softmax across 64 keys: rows live across 16-lane halves
        float p[4][8];
#pragma unroll
        for (int r = 0; r < 8; ++r) {
            float v0 = s[0][r] * scale, v1 = s[1][r] * scale;
            float v2 = s[2][r] * scale, v3 = s[3][r] * scale;
            float mx = fmaxf(fmaxf(v0, v1), fmaxf(v2, v3));
#pragma unroll
            for (int m = 1; m < 16; m <<= 1) mx = fmaxf(mx, __shfl_xor(mx, m));
            float mn    = fmaxf(mi[r], mx);
            float alpha = __expf(mi[r] - mn);
            float e0 = __expf(v0 - mn), e1 = __expf(v1 - mn);
            float e2 = __expf(v2 - mn), e3 = __expf(v3 - mn);
            p[0][r] = e0; p[1][r] = e1; p[2][r] = e2; p[3][r] = e3;
            float rs = (e0 + e1) + (e2 + e3);
#pragma unroll
            for (int m = 1; m < 16; m <<= 1) rs += __shfl_xor(rs, m);
            li[r] = li[r] * alpha + rs;
            mi[r] = mn;
#pragma unroll
            for (int nt = 0; nt < 4; ++nt) oacc[nt][r] *= alpha;
        }

        // stage P (16x64 f16) through LDS to reshape into A-fragments
#pragma unroll
        for (int r = 0; r < 8; ++r) {
            int row = r + 8 * lh;
#pragma unroll
            for (int nt = 0; nt < 4; ++nt)
                Plds[wave * 16 * KS + row * KS + nt * 16 + ln16] = (_Float16)p[nt][r];
        }
        __syncthreads();

        Frag pf0, pf1;
#pragma unroll
        for (int i = 0; i < 8; ++i) {
            pf0.u[i] = *(const unsigned int*)(&Plds[wave * 16 * KS + ln16 * KS + kofs[i]]);
            pf1.u[i] = *(const unsigned int*)(&Plds[wave * 16 * KS + ln16 * KS + 32 + kofs[i]]);
        }
#pragma unroll
        for (int nt = 0; nt < 4; ++nt) {
            Frag vf0, vf1;
#pragma unroll
            for (int i = 0; i < 8; ++i) {
                vf0.u[i] = *(const unsigned int*)(&Vlds[(nt * 16 + ln16) * KS + kofs[i]]);
                vf1.u[i] = *(const unsigned int*)(&Vlds[(nt * 16 + ln16) * KS + 32 + kofs[i]]);
            }
            oacc[nt] = __builtin_amdgcn_wmma_f32_16x16x32_f16(false, pf0.h, false, vf0.h,
                                                              (short)0, oacc[nt], false, false);
            oacc[nt] = __builtin_amdgcn_wmma_f32_16x16x32_f16(false, pf1.h, false, vf1.h,
                                                              (short)0, oacc[nt], false, false);
        }
    }

    // normalize and write O
#pragma unroll
    for (int nt = 0; nt < 4; ++nt)
#pragma unroll
        for (int r = 0; r < 8; ++r) {
            float v = oacc[nt][r] / li[r];
            O[base + (size_t)(q0 + wave * 16 + r + 8 * lh) * D + nt * 16 + ln16] = (_Float16)v;
        }
}

// ---------------------------------------------------------------------------
// LayerNorm over D=512, one block (256 thr) per row; writes f32 + f16 copies
// ---------------------------------------------------------------------------
__global__ __launch_bounds__(256) void layernorm(const float* __restrict__ in,
                                                 const float* __restrict__ g,
                                                 const float* __restrict__ be,
                                                 float* __restrict__ xf,
                                                 _Float16* __restrict__ xh) {
    const int D = 512;
    __shared__ float sm[8], sv[8];
    const int tid = threadIdx.x;
    const size_t row = blockIdx.x;
    const float* p = in + row * D;
    float a = p[tid], b2 = p[tid + 256];

    float s = a + b2;
#pragma unroll
    for (int m = 16; m >= 1; m >>= 1) s += __shfl_xor(s, m);
    if ((tid & 31) == 0) sm[tid >> 5] = s;
    __syncthreads();
    float mean = 0.f;
#pragma unroll
    for (int i = 0; i < 8; ++i) mean += sm[i];
    mean *= (1.f / 512.f);

    float da = a - mean, db = b2 - mean;
    float v = da * da + db * db;
#pragma unroll
    for (int m = 16; m >= 1; m >>= 1) v += __shfl_xor(v, m);
    if ((tid & 31) == 0) sv[tid >> 5] = v;
    __syncthreads();
    float var = 0.f;
#pragma unroll
    for (int i = 0; i < 8; ++i) var += sv[i];
    float rstd = rsqrtf(var * (1.f / 512.f) + 1e-5f);

    float o1 = da * rstd * g[tid] + be[tid];
    float o2 = db * rstd * g[tid + 256] + be[tid + 256];
    xf[row * D + tid]       = o1;
    xf[row * D + tid + 256] = o2;
    xh[row * D + tid]       = (_Float16)o1;
    xh[row * D + tid + 256] = (_Float16)o2;
}

// ---------------------------------------------------------------------------
// output projection: out[M,20] = x @ W_out + b_out
// ---------------------------------------------------------------------------
__global__ __launch_bounds__(256) void out_proj(const float* __restrict__ x,
                                                const float* __restrict__ Wo,
                                                const float* __restrict__ bo,
                                                float* __restrict__ out,
                                                int M, int D, int OUT) {
    int idx = blockIdx.x * 256 + threadIdx.x;
    if (idx >= M * OUT) return;
    int row = idx / OUT, o = idx - row * OUT;
    float s = bo[o];
    const float* xr = x + (size_t)row * D;
    for (int k = 0; k < D; ++k) s += xr[k] * Wo[(size_t)k * OUT + o];
    out[idx] = s;
}

// ---------------------------------------------------------------------------
extern "C" void kernel_launch(void* const* d_in, const int* in_sizes, int n_in,
                              void* d_out, int out_size, void* d_ws, size_t ws_size,
                              hipStream_t stream) {
    const int B = 4, S = 2048, D = 512, F = 2048, OUT = 20, L = 6, H = 8;
    const int M = B * S;

    const float* src   = (const float*)d_in[0];
    const float* W_in  = (const float*)d_in[1];
    const float* b_in  = (const float*)d_in[2];
    const float* Wq    = (const float*)d_in[3];
    const float* bq    = (const float*)d_in[4];
    const float* Wk    = (const float*)d_in[5];
    const float* bk    = (const float*)d_in[6];
    const float* Wv    = (const float*)d_in[7];
    const float* bv    = (const float*)d_in[8];
    const float* Wo    = (const float*)d_in[9];
    const float* bo    = (const float*)d_in[10];
    const float* ln1g  = (const float*)d_in[11];
    const float* ln1b  = (const float*)d_in[12];
    const float* W1    = (const float*)d_in[13];
    const float* b1    = (const float*)d_in[14];
    const float* W2    = (const float*)d_in[15];
    const float* b2    = (const float*)d_in[16];
    const float* ln2g  = (const float*)d_in[17];
    const float* ln2b  = (const float*)d_in[18];
    const float* W_out = (const float*)d_in[19];
    const float* b_out = (const float*)d_in[20];

    char* ws = (char*)d_ws;
    auto alloc = [&](size_t bytes) -> char* {
        char* p = ws;
        ws += (bytes + 255) & ~(size_t)255;
        return p;
    };

    float*     x_f32 = (float*)alloc((size_t)M * D * 4);
    float*     tmp   = (float*)alloc((size_t)M * D * 4);
    _Float16*  x_h   = (_Float16*)alloc((size_t)M * D * 2);
    _Float16*  qb    = (_Float16*)alloc((size_t)M * D * 2);
    _Float16*  kb    = (_Float16*)alloc((size_t)M * D * 2);
    _Float16*  vb    = (_Float16*)alloc((size_t)M * D * 2);
    _Float16*  ob    = (_Float16*)alloc((size_t)M * D * 2);
    _Float16*  h1    = (_Float16*)alloc((size_t)M * F * 2);
    _Float16*  Wq16  = (_Float16*)alloc((size_t)L * D * D * 2);
    _Float16*  Wk16  = (_Float16*)alloc((size_t)L * D * D * 2);
    _Float16*  Wv16  = (_Float16*)alloc((size_t)L * D * D * 2);
    _Float16*  Wo16  = (_Float16*)alloc((size_t)L * D * D * 2);
    _Float16*  W116  = (_Float16*)alloc((size_t)L * D * F * 2);
    _Float16*  W216  = (_Float16*)alloc((size_t)L * F * D * 2);

    // weight conversions (every call; deterministic)
    {
        int nqkv = L * D * D, nff = L * D * F;
        cvt_f32_f16<<<(nqkv + 255) / 256, 256, 0, stream>>>(Wq, Wq16, nqkv);
        cvt_f32_f16<<<(nqkv + 255) / 256, 256, 0, stream>>>(Wk, Wk16, nqkv);
        cvt_f32_f16<<<(nqkv + 255) / 256, 256, 0, stream>>>(Wv, Wv16, nqkv);
        cvt_f32_f16<<<(nqkv + 255) / 256, 256, 0, stream>>>(Wo, Wo16, nqkv);
        cvt_f32_f16<<<(nff + 255) / 256, 256, 0, stream>>>(W1, W116, nff);
        cvt_f32_f16<<<(nff + 255) / 256, 256, 0, stream>>>(W2, W216, nff);
    }

    input_proj<<<(M * D + 255) / 256, 256, 0, stream>>>(src, W_in, b_in, x_f32, x_h, M, D);

    const dim3 gD(D / 64, M / 256);   // N=512 GEMMs
    const dim3 gF(F / 64, M / 256);   // N=2048 GEMM
    for (int l = 0; l < L; ++l) {
        const size_t wqo = (size_t)l * D * D;
        const size_t wfo = (size_t)l * D * F;
        // QKV
        gemm_wmma<0><<<gD, 256, 0, stream>>>(x_h, Wq16 + wqo, bq + l * D, nullptr, qb, M, D, D);
        gemm_wmma<0><<<gD, 256, 0, stream>>>(x_h, Wk16 + wqo, bk + l * D, nullptr, kb, M, D, D);
        gemm_wmma<0><<<gD, 256, 0, stream>>>(x_h, Wv16 + wqo, bv + l * D, nullptr, vb, M, D, D);
        // attention
        flash_attn<<<dim3(S / 64, H, B), 128, 0, stream>>>(qb, kb, vb, ob, S, D);
        // output proj + residual
        gemm_wmma<2><<<gD, 256, 0, stream>>>(ob, Wo16 + wqo, bo + l * D, x_f32, tmp, M, D, D);
        layernorm<<<M, 256, 0, stream>>>(tmp, ln1g + l * D, ln1b + l * D, x_f32, x_h);
        // FFN
        gemm_wmma<1><<<gF, 256, 0, stream>>>(x_h, W116 + wfo, b1 + l * F, nullptr, h1, M, F, D);
        gemm_wmma<2><<<gD, 256, 0, stream>>>(h1, W216 + wfo, b2 + l * D, x_f32, tmp, M, D, F);
        layernorm<<<M, 256, 0, stream>>>(tmp, ln2g + l * D, ln2b + l * D, x_f32, x_h);
    }

    out_proj<<<(M * OUT + 255) / 256, 256, 0, stream>>>(x_f32, W_out, b_out, (float*)d_out, M, D, OUT);
}
